// Model_10574209483563
// MI455X (gfx1250) — compile-verified, matrix operand order checked by reading, not run on hardware
//
#include <hip/hip_runtime.h>
#include <hip/hip_bf16.h>

typedef __attribute__((ext_vector_type(16))) _Float16 v16h;
typedef __attribute__((ext_vector_type(8)))  _Float16 v8h;
typedef __attribute__((ext_vector_type(8)))  float    v8f;

__device__ __forceinline__ v8f vzero() { v8f z = {0.f,0.f,0.f,0.f,0.f,0.f,0.f,0.f}; return z; }
__device__ __forceinline__ float sigf(float x) { return 1.f / (1.f + __expf(-x)); }

// Load one 16x32 (MxK) f16 fragment per the CDNA5 ISA 16-bit A/B layout:
// lane L: row = row0 + (L&15); halfs 0..7 <- K = k0 + (L>>4)*8 + 0..7,
//                              halfs 8..15 <- K = k0 + 16 + (L>>4)*8 + 0..7.
// For B (K x N, B[k][n] = W[n][k] with W row-major [N][K]) the same loader
// applies with row0 = n0.
__device__ __forceinline__ v16h load_frag(const _Float16* p, int ld, int row0, int k0, int lane) {
  const _Float16* r = p + (size_t)(row0 + (lane & 15)) * ld + k0 + ((lane >> 4) << 3);
  v8h a = *(const v8h*)r;
  v8h b = *(const v8h*)(r + 16);
  v16h o;
#pragma unroll
  for (int i = 0; i < 8; ++i) { o[i] = a[i]; o[8 + i] = b[i]; }
  return o;
}

__device__ __forceinline__ v8f wmma32(v16h a, v16h b, v8f c) {
  return __builtin_amdgcn_wmma_f32_16x16x32_f16(false, a, false, b, (short)0, c, false, false);
}

// ---------------------------------------------------------------- utilities
__global__ void k_f32_to_f16(const float* __restrict__ s, _Float16* __restrict__ d, int n) {
  int i = blockIdx.x * blockDim.x + threadIdx.x;
  if (i < n) d[i] = (_Float16)s[i];
}

// pos_emb -> xenc[w][b][0..31]
__global__ void k_posemb(const float* __restrict__ pos, const float* __restrict__ normp,
                         const float* __restrict__ pw, const float* __restrict__ pb,
                         _Float16* __restrict__ xenc) {
  int i = blockIdx.x * blockDim.x + threadIdx.x;   // 32*256*32
  if (i >= 32 * 256 * 32) return;
  int p = i & 31;
  int r = i >> 5;                                  // r = b*256 + w
  int b = r >> 8, w = r & 255;
  const float* q = pos + (size_t)r * 4;
  float s = pb[p];
#pragma unroll
  for (int k = 0; k < 4; ++k) s += (q[k] / normp[k]) * pw[p * 4 + k];
  xenc[((size_t)w * 32 + b) * 288 + p] = (_Float16)sigf(s);
}

// -------------------------------------------------- Stage A: char-level LSTM
// 256 blocks x 512 threads; block handles 32 word-rows, 16 waves: wave w owns
// hidden slice j0=w*16 and gate tiles {j0, 256+j0, 512+j0, 768+j0}.
__global__ __launch_bounds__(512) void k_char_lstm(
    const int* __restrict__ words,       // (8192,16)
    const _Float16* __restrict__ ce16,   // (128,256)
    const _Float16* __restrict__ wih16,  // (1024,256)
    const _Float16* __restrict__ whh16,  // (1024,256)
    const float* __restrict__ bih, const float* __restrict__ bhh,
    _Float16* __restrict__ xenc)         // (256,32,288), word part at +32
{
  __shared__ _Float16 xs[32][256];
  __shared__ _Float16 hs[32][256];
  const int tid = threadIdx.x;
  const int lane = tid & 31;
  const int wv = tid >> 5;
  const int hi = lane >> 4;
  const int ncol = lane & 15;
  const int j0 = wv * 16;
  const int rowbase = blockIdx.x * 32;

  float bsum[4];
#pragma unroll
  for (int g = 0; g < 4; ++g) {
    int n = g * 256 + j0 + ncol;
    bsum[g] = bih[n] + bhh[n];
  }

  v8f cst[2];
  cst[0] = vzero(); cst[1] = vzero();

#pragma unroll 1
  for (int t = 0; t < 16; ++t) {
    __syncthreads();
    // gather char embeddings for this timestep into LDS (f16)
#pragma unroll 1
    for (int c = tid; c < 32 * 32; c += 512) {
      int r = c >> 5, ch = c & 31;
      int idx = words[(size_t)(rowbase + r) * 16 + t];
      *(v8h*)&xs[r][ch * 8] = *(const v8h*)(ce16 + (size_t)idx * 256 + ch * 8);
    }
    __syncthreads();

    v8f acc[2][4];
#pragma unroll
    for (int mt = 0; mt < 2; ++mt)
#pragma unroll
      for (int g = 0; g < 4; ++g) acc[mt][g] = vzero();

#pragma unroll 1
    for (int kk = 0; kk < 256; kk += 32) {
      v16h a0 = load_frag(&xs[0][0], 256, 0,  kk, lane);
      v16h a1 = load_frag(&xs[0][0], 256, 16, kk, lane);
#pragma unroll
      for (int g = 0; g < 4; ++g) {
        v16h b = load_frag(wih16, 256, g * 256 + j0, kk, lane);
        acc[0][g] = wmma32(a0, b, acc[0][g]);
        acc[1][g] = wmma32(a1, b, acc[1][g]);
      }
    }
    if (t > 0) {
#pragma unroll 1
      for (int kk = 0; kk < 256; kk += 32) {
        v16h a0 = load_frag(&hs[0][0], 256, 0,  kk, lane);
        v16h a1 = load_frag(&hs[0][0], 256, 16, kk, lane);
#pragma unroll
        for (int g = 0; g < 4; ++g) {
          v16h b = load_frag(whh16, 256, g * 256 + j0, kk, lane);
          acc[0][g] = wmma32(a0, b, acc[0][g]);
          acc[1][g] = wmma32(a1, b, acc[1][g]);
        }
      }
    }
    __syncthreads();
#pragma unroll
    for (int mt = 0; mt < 2; ++mt)
#pragma unroll
      for (int i = 0; i < 8; ++i) {
        float gi = acc[mt][0][i] + bsum[0];
        float gf = acc[mt][1][i] + bsum[1];
        float gg = acc[mt][2][i] + bsum[2];
        float go = acc[mt][3][i] + bsum[3];
        float c = sigf(gf) * cst[mt][i] + sigf(gi) * tanhf(gg);
        cst[mt][i] = c;
        float h = sigf(go) * tanhf(c);
        hs[mt * 16 + i + 8 * hi][j0 + ncol] = (_Float16)h;
      }
  }
  __syncthreads();
  // final h -> xenc[w][b][32+j]
#pragma unroll 1
  for (int q = tid; q < 32 * 256; q += 512) {
    int r = q >> 8, j = q & 255;
    int gr = rowbase + r;
    int b = gr >> 8, w = gr & 255;
    xenc[((size_t)w * 32 + b) * 288 + 32 + j] = hs[r][j];
  }
}

// -------------------------------------------------- Stage C: bi-LSTM encoder
// grid=2 (dir), 1024 threads = 32 waves; wave: j0=(wv&15)*16, mt=wv>>4.
__global__ __launch_bounds__(1024) void k_encoder(
    const _Float16* __restrict__ xenc,     // (256,32,288)
    const _Float16* __restrict__ fwih, const _Float16* __restrict__ fwhh,
    const float* __restrict__ fbih, const float* __restrict__ fbhh,
    const _Float16* __restrict__ bwih, const _Float16* __restrict__ bwhh,
    const float* __restrict__ bbih, const float* __restrict__ bbhh,
    _Float16* __restrict__ tok,            // (32,256,512)
    float* __restrict__ hfin)              // (2,32,256)
{
  const int dir = blockIdx.x;
  const _Float16* wih16 = dir ? bwih : fwih;
  const _Float16* whh16 = dir ? bwhh : fwhh;
  const float* bihp = dir ? bbih : fbih;
  const float* bhhp = dir ? bbhh : fbhh;

  __shared__ _Float16 hsh[32][256];
  const int tid = threadIdx.x;
  const int lane = tid & 31;
  const int wv = tid >> 5;
  const int hi = lane >> 4;
  const int ncol = lane & 15;
  const int j0 = (wv & 15) * 16;
  const int mt = wv >> 4;

  float bsum[4];
#pragma unroll
  for (int g = 0; g < 4; ++g) {
    int n = g * 256 + j0 + ncol;
    bsum[g] = bihp[n] + bhhp[n];
  }
  v8f cst = vzero();

#pragma unroll 1
  for (int s = 0; s < 256; ++s) {
    const int ww = dir ? (255 - s) : s;
    const _Float16* xrow = xenc + (size_t)ww * 32 * 288;
    v8f acc[4];
#pragma unroll
    for (int g = 0; g < 4; ++g) acc[g] = vzero();
#pragma unroll 1
    for (int kk = 0; kk < 288; kk += 32) {
      v16h a = load_frag(xrow, 288, mt * 16, kk, lane);
#pragma unroll
      for (int g = 0; g < 4; ++g) {
        v16h b = load_frag(wih16, 288, g * 256 + j0, kk, lane);
        acc[g] = wmma32(a, b, acc[g]);
      }
    }
    if (s > 0) {
#pragma unroll 1
      for (int kk = 0; kk < 256; kk += 32) {
        v16h a = load_frag(&hsh[0][0], 256, mt * 16, kk, lane);
#pragma unroll
        for (int g = 0; g < 4; ++g) {
          v16h b = load_frag(whh16, 256, g * 256 + j0, kk, lane);
          acc[g] = wmma32(a, b, acc[g]);
        }
      }
    }
    __syncthreads();
#pragma unroll
    for (int i = 0; i < 8; ++i) {
      float gi = acc[0][i] + bsum[0];
      float gf = acc[1][i] + bsum[1];
      float gg = acc[2][i] + bsum[2];
      float go = acc[3][i] + bsum[3];
      float c = sigf(gf) * cst[i] + sigf(gi) * tanhf(gg);
      cst[i] = c;
      float h = sigf(go) * tanhf(c);
      int b = mt * 16 + i + 8 * hi;
      int j = j0 + ncol;
      hsh[b][j] = (_Float16)h;
      tok[((size_t)b * 256 + ww) * 512 + dir * 256 + j] = (_Float16)h;
      if (s == 255) hfin[(size_t)dir * 8192 + (size_t)b * 256 + j] = h;
    }
    __syncthreads();
  }
}

// enc_doc = stack([hf,hb]).reshape(32,512) -> initial dec_in (flat copy)
__global__ void k_decin_init(const float* __restrict__ hfin, _Float16* __restrict__ dec_in) {
  int i = blockIdx.x * blockDim.x + threadIdx.x;
  if (i < 16384) dec_in[i] = (_Float16)hfin[i];
}

// ---------------------------------------- Stage E: enc_proj = tok @ encw_w.T
__global__ __launch_bounds__(512) void k_encproj(
    const _Float16* __restrict__ tok,   // (8192,512)
    const _Float16* __restrict__ w16,   // (256,512)
    const float* __restrict__ bias,     // (256)
    float* __restrict__ out)            // (8192,256)
{
  const int tid = threadIdx.x;
  const int lane = tid & 31;
  const int wv = tid >> 5;
  const int hi = lane >> 4;
  const int ncol = lane & 15;
  const int n0 = wv * 16;
  const int rowbase = blockIdx.x * 32;

  v8f acc[2];
  acc[0] = vzero(); acc[1] = vzero();
#pragma unroll 1
  for (int kk = 0; kk < 512; kk += 32) {
    v16h a0 = load_frag(tok, 512, rowbase,      kk, lane);
    v16h a1 = load_frag(tok, 512, rowbase + 16, kk, lane);
    v16h b  = load_frag(w16, 512, n0, kk, lane);
    acc[0] = wmma32(a0, b, acc[0]);
    acc[1] = wmma32(a1, b, acc[1]);
  }
  float bv = bias[n0 + ncol];
#pragma unroll
  for (int mt = 0; mt < 2; ++mt)
#pragma unroll
    for (int i = 0; i < 8; ++i) {
      int r = rowbase + mt * 16 + i + 8 * hi;
      out[(size_t)r * 256 + n0 + ncol] = acc[mt][i] + bv;
    }
}

// ---------------------------------------- Decoder step 1: cell (h=c=0) + dq
__global__ __launch_bounds__(512) void k_dec_cell(
    const _Float16* __restrict__ dec_in,  // (32,512)
    const _Float16* __restrict__ cwih16,  // (1024,512)
    const float* __restrict__ cbih, const float* __restrict__ cbhh,
    const _Float16* __restrict__ decw16,  // (256,256)
    const float* __restrict__ decb,
    float* __restrict__ dq)               // (32,256)
{
  __shared__ _Float16 hx[32][256];
  const int tid = threadIdx.x;
  const int lane = tid & 31;
  const int wv = tid >> 5;
  const int hi = lane >> 4;
  const int ncol = lane & 15;
  const int j0 = wv * 16;

  v8f acc[2][4];
#pragma unroll
  for (int mt = 0; mt < 2; ++mt)
#pragma unroll
    for (int g = 0; g < 4; ++g) acc[mt][g] = vzero();
#pragma unroll 1
  for (int kk = 0; kk < 512; kk += 32) {
    v16h a0 = load_frag(dec_in, 512, 0,  kk, lane);
    v16h a1 = load_frag(dec_in, 512, 16, kk, lane);
#pragma unroll
    for (int g = 0; g < 4; ++g) {
      v16h b = load_frag(cwih16, 512, g * 256 + j0, kk, lane);
      acc[0][g] = wmma32(a0, b, acc[0][g]);
      acc[1][g] = wmma32(a1, b, acc[1][g]);
    }
  }
  float bsum[4];
#pragma unroll
  for (int g = 0; g < 4; ++g) {
    int n = g * 256 + j0 + ncol;
    bsum[g] = cbih[n] + cbhh[n];
  }
#pragma unroll
  for (int mt = 0; mt < 2; ++mt)
#pragma unroll
    for (int i = 0; i < 8; ++i) {
      float gi = acc[mt][0][i] + bsum[0];
      float gg = acc[mt][2][i] + bsum[2];
      float go = acc[mt][3][i] + bsum[3];
      float c = sigf(gi) * tanhf(gg);            // f-gate drops: c_prev = 0
      float h = sigf(go) * tanhf(c);
      hx[mt * 16 + i + 8 * hi][j0 + ncol] = (_Float16)h;
    }
  __syncthreads();
  // dq = hx @ dec_w.T + dec_b  (wave wv -> n-tile j0)
  v8f a2[2];
  a2[0] = vzero(); a2[1] = vzero();
#pragma unroll 1
  for (int kk = 0; kk < 256; kk += 32) {
    v16h a0 = load_frag(&hx[0][0], 256, 0,  kk, lane);
    v16h a1 = load_frag(&hx[0][0], 256, 16, kk, lane);
    v16h b  = load_frag(decw16, 256, j0, kk, lane);
    a2[0] = wmma32(a0, b, a2[0]);
    a2[1] = wmma32(a1, b, a2[1]);
  }
  float bv = decb[j0 + ncol];
#pragma unroll
  for (int mt = 0; mt < 2; ++mt)
#pragma unroll
    for (int i = 0; i < 8; ++i)
      dq[(size_t)(mt * 16 + i + 8 * hi) * 256 + j0 + ncol] = a2[mt][i] + bv;
}

// -------- Decoder step 2: scores + log_softmax + argmax + dec_in gather
__global__ __launch_bounds__(256) void k_dec_score(
    const float* __restrict__ dq,       // (32,256)
    const float* __restrict__ encproj,  // (8192,256)
    const float* __restrict__ refw, const float* __restrict__ refb,
    const _Float16* __restrict__ tok,   // (8192,512)
    _Float16* __restrict__ dec_in,      // (32,512)
    float* __restrict__ out_probs,      // (32,64,256)
    float* __restrict__ out_peaks,      // (32,64)
    int t)
{
  const int b = blockIdx.x;
  const int tid = threadIdx.x;
  __shared__ float sdq[256], srw[256], red[256];
  __shared__ int ridx[256];
  sdq[tid] = dq[(size_t)b * 256 + tid];
  srw[tid] = refw[tid];
  __syncthreads();

  const float4* ep = (const float4*)(encproj + ((size_t)b * 256 + tid) * 256);
  float s = refb[0];
#pragma unroll 4
  for (int e4 = 0; e4 < 64; ++e4) {
    float4 v = ep[e4];
    int e = e4 * 4;
    s += srw[e]     * tanhf(sdq[e]     + v.x)
       + srw[e + 1] * tanhf(sdq[e + 1] + v.y)
       + srw[e + 2] * tanhf(sdq[e + 2] + v.z)
       + srw[e + 3] * tanhf(sdq[e + 3] + v.w);
  }
  // argmax (first occurrence) + max
  red[tid] = s; ridx[tid] = tid;
  __syncthreads();
  for (int off = 128; off > 0; off >>= 1) {
    if (tid < off) {
      float a = red[tid], c = red[tid + off];
      int ia = ridx[tid], ic = ridx[tid + off];
      if (c > a || (c == a && ic < ia)) { red[tid] = c; ridx[tid] = ic; }
    }
    __syncthreads();
  }
  float m = red[0];
  int peak = ridx[0];
  __syncthreads();
  red[tid] = __expf(s - m);
  __syncthreads();
  for (int off = 128; off > 0; off >>= 1) {
    if (tid < off) red[tid] += red[tid + off];
    __syncthreads();
  }
  float lse = __logf(red[0]);
  out_probs[((size_t)b * 64 + t) * 256 + tid] = s - m - lse;
  if (tid == 0) out_peaks[(size_t)b * 64 + t] = (float)peak;
  // next dec_in = tok[b, peak]
  const _Float16* tr = tok + ((size_t)b * 256 + peak) * 512;
  dec_in[(size_t)b * 512 + tid] = tr[tid];
  dec_in[(size_t)b * 512 + 256 + tid] = tr[256 + tid];
}

// ---------------------------------------------------------------- launcher
extern "C" void kernel_launch(void* const* d_in, const int* in_sizes, int n_in,
                              void* d_out, int out_size, void* d_ws, size_t ws_size,
                              hipStream_t stream) {
  (void)in_sizes; (void)n_in; (void)out_size; (void)ws_size;
  const int*   words     = (const int*)  d_in[0];
  const float* positions = (const float*)d_in[1];
  const float* char_emb  = (const float*)d_in[2];
  const float* wl_wih = (const float*)d_in[3];
  const float* wl_whh = (const float*)d_in[4];
  const float* wl_bih = (const float*)d_in[5];
  const float* wl_bhh = (const float*)d_in[6];
  const float* norm_pos = (const float*)d_in[7];
  const float* pos_w = (const float*)d_in[8];
  const float* pos_b = (const float*)d_in[9];
  const float* ef_wih = (const float*)d_in[10];
  const float* ef_whh = (const float*)d_in[11];
  const float* ef_bih = (const float*)d_in[12];
  const float* ef_bhh = (const float*)d_in[13];
  const float* eb_wih = (const float*)d_in[14];
  const float* eb_whh = (const float*)d_in[15];
  const float* eb_bih = (const float*)d_in[16];
  const float* eb_bhh = (const float*)d_in[17];
  const float* cell_wih = (const float*)d_in[18];
  const float* cell_bih = (const float*)d_in[20];
  const float* cell_bhh = (const float*)d_in[21];
  const float* dec_w = (const float*)d_in[22];
  const float* dec_b = (const float*)d_in[23];
  const float* encw_w = (const float*)d_in[24];
  const float* encw_b = (const float*)d_in[25];
  const float* ref_w = (const float*)d_in[26];
  const float* ref_b = (const float*)d_in[27];

  char* ws = (char*)d_ws;
  size_t off = 0;
  auto alloc = [&](size_t bytes) { size_t r = off; off = (off + bytes + 255) & ~(size_t)255; return r; };
  _Float16* ce16    = (_Float16*)(ws + alloc(32768   * 2));
  _Float16* wlwih16 = (_Float16*)(ws + alloc(262144  * 2));
  _Float16* wlwhh16 = (_Float16*)(ws + alloc(262144  * 2));
  _Float16* efwih16 = (_Float16*)(ws + alloc(294912  * 2));
  _Float16* efwhh16 = (_Float16*)(ws + alloc(262144  * 2));
  _Float16* ebwih16 = (_Float16*)(ws + alloc(294912  * 2));
  _Float16* ebwhh16 = (_Float16*)(ws + alloc(262144  * 2));
  _Float16* cwih16  = (_Float16*)(ws + alloc(524288  * 2));
  _Float16* decw16  = (_Float16*)(ws + alloc(65536   * 2));
  _Float16* encw16  = (_Float16*)(ws + alloc(131072  * 2));
  _Float16* xenc    = (_Float16*)(ws + alloc(2359296 * 2)); // (256,32,288)
  _Float16* tok     = (_Float16*)(ws + alloc(4194304 * 2)); // (32,256,512)
  float*    hfin    = (float*)   (ws + alloc(16384   * 4)); // (2,32,256)
  _Float16* dec_in  = (_Float16*)(ws + alloc(16384   * 2)); // (32,512)
  float*    encproj = (float*)   (ws + alloc(2097152 * 4)); // (8192,256)
  float*    dq      = (float*)   (ws + alloc(8192    * 4)); // (32,256)

  auto cvt = [&](const float* s, _Float16* d, int n) {
    k_f32_to_f16<<<(n + 255) / 256, 256, 0, stream>>>(s, d, n);
  };
  cvt(char_emb, ce16,    32768);
  cvt(wl_wih,   wlwih16, 262144);
  cvt(wl_whh,   wlwhh16, 262144);
  cvt(ef_wih,   efwih16, 294912);
  cvt(ef_whh,   efwhh16, 262144);
  cvt(eb_wih,   ebwih16, 294912);
  cvt(eb_whh,   ebwhh16, 262144);
  cvt(cell_wih, cwih16,  524288);
  cvt(dec_w,    decw16,  65536);
  cvt(encw_w,   encw16,  131072);

  k_posemb<<<(32 * 256 * 32 + 255) / 256, 256, 0, stream>>>(positions, norm_pos, pos_w, pos_b, xenc);

  k_char_lstm<<<256, 512, 0, stream>>>(words, ce16, wlwih16, wlwhh16, wl_bih, wl_bhh, xenc);

  k_encoder<<<2, 1024, 0, stream>>>(xenc, efwih16, efwhh16, ef_bih, ef_bhh,
                                    ebwih16, ebwhh16, eb_bih, eb_bhh, tok, hfin);

  k_decin_init<<<64, 256, 0, stream>>>(hfin, dec_in);

  k_encproj<<<256, 512, 0, stream>>>(tok, encw16, encw_b, encproj);

  float* out_probs = (float*)d_out;            // (32,64,256)
  float* out_peaks = (float*)d_out + 524288;   // (32,64)
  for (int t = 0; t < 64; ++t) {
    k_dec_cell<<<1, 512, 0, stream>>>(dec_in, cwih16, cell_bih, cell_bhh, decw16, dec_b, dq);
    k_dec_score<<<32, 256, 0, stream>>>(dq, encproj, ref_w, ref_b, tok, dec_in,
                                        out_probs, out_peaks, t);
  }
}
